// MultiLayerLSTM_90082644066693
// MI455X (gfx1250) — matrix operand order repaired and checked by reading, not compiled, not run on hardware
//
#include <hip/hip_runtime.h>

// ---------------- problem constants ----------------
#define B_    64
#define T_    1024
#define H_    512
#define L_    3
#define G4H   2048            // 4*H
#define M_    (B_ * T_)       // 65536 rows of the projection GEMM
#define NWG_REC 32            // recurrent kernel: 32 WGs x 4 waves = 128 waves
#define REC_TPB 128

typedef __attribute__((ext_vector_type(16))) __bf16 v16bf;
typedef __attribute__((ext_vector_type(8)))  float  v8f;

union Frag { v16bf v; uint4 q[2]; };

__device__ __forceinline__ unsigned short f2bf(float f) {
    unsigned int u = __float_as_uint(f);
    u += 0x7fffu + ((u >> 16) & 1u);          // round-to-nearest-even
    return (unsigned short)(u >> 16);
}
__device__ __forceinline__ float sigmoidf_(float x) { return 1.0f / (1.0f + __expf(-x)); }

// ---- CDNA5 async global->LDS copy (ASYNCcnt-tracked) ----
__device__ __forceinline__ void async_g2l_b128(unsigned lds_addr, const void* gptr) {
    asm volatile("global_load_async_to_lds_b128 %0, %1, off"
                 :: "v"(lds_addr), "v"((unsigned long long)(uintptr_t)gptr)
                 : "memory");
}
__device__ __forceinline__ void wait_async0() {
    asm volatile("s_wait_asynccnt 0x0" ::: "memory");
}

// ---------------- setup kernels ----------------
__global__ void k_convert_bf16(const float* __restrict__ src,
                               unsigned short* __restrict__ dst, int n) {
    int i = blockIdx.x * blockDim.x + threadIdx.x;
    int stride = gridDim.x * blockDim.x;
    for (; i < n; i += stride) dst[i] = f2bf(src[i]);
}

__global__ void k_bias_bar(const float* __restrict__ bih, const float* __restrict__ bhh,
                           float* __restrict__ bc, int nb, unsigned int* __restrict__ bar) {
    int i = blockIdx.x * blockDim.x + threadIdx.x;
    int stride = gridDim.x * blockDim.x;
    for (int j = i; j < nb; j += stride) bc[j] = bih[j] + bhh[j];
    if (i == 0) *bar = 0u;
}

// ---------------- input-projection GEMM: xg = x * Wih^T + (bih+bhh) ----------------
// A: [M,512] bf16 row-major. B is Wih [2048,512] bf16 row-major (N-major of Wih^T).
// Block: 8 waves, 128(M) x 64(N) tile. Per iteration a 64-deep K slab of B (8KB)
// is async-staged into LDS (double-buffered) and shared by all 8 waves; A is
// register double-buffered per wave. 8 WMMAs between successive barriers.
__global__ __launch_bounds__(256, 1)
void k_gemm_xproj(const unsigned short* __restrict__ A,
                  const unsigned short* __restrict__ Wn,
                  const float* __restrict__ bias,
                  float* __restrict__ Xg) {
    __shared__ unsigned short sB[2][2 * 4 * 32 * 16];   // [buf][chunk*2048 + (j*32+lane)*16]

    const int tid  = threadIdx.x;
    const int lane = tid & 31;
    const int wave = tid >> 5;
    const int m0   = blockIdx.x * 128 + wave * 16;
    const int n0   = blockIdx.y * 64;
    const int l15  = lane & 15;
    const int hi   = lane >> 4;
    const int akoff = hi ? 8 : 0;     // A: lanes16-31 hold K+8 / K+24
    // B: lanes16-31 hold K+8 / K+24 (handled inside bSrcBase)

    // this thread's slice of the cooperative B stage: 16B per 32-K chunk
    const int qj    = tid >> 6;          // 0..3 : which 16-col slice
    const int qlane = (tid & 63) >> 1;   // 0..31: destination lane of the fragment
    const int qhalf = tid & 1;           // 0/1  : low/high 16B of the 32B fragment
    const unsigned short* bSrcBase =
        Wn + (size_t)(n0 + qj * 16 + (qlane & 15)) * H_ + ((qlane < 16) ? 0 : 8) + qhalf * 16;
    const unsigned ldsOffThread = (unsigned)(tid * 16);  // bytes within one 4KB chunk

    auto stage64 = [&](int buf, int k0) {                // stage k0..k0+64
        unsigned lds = (unsigned)(uintptr_t)(&sB[buf][0]) + ldsOffThread;
        async_g2l_b128(lds,        bSrcBase + k0);
        async_g2l_b128(lds + 4096, bSrcBase + k0 + 32);
    };

    v8f acc[4];
#pragma unroll
    for (int j = 0; j < 4; ++j) {
        float bv = bias[n0 + 16 * j + l15];   // bias depends only on N (= lane)
#pragma unroll
        for (int i = 0; i < 8; ++i) acc[j][i] = bv;
    }

    const unsigned short* aRow = A + (size_t)(m0 + l15) * H_ + akoff;

    // prologue: stage B for k=[0,64), preload A fragments for k=[0,64)
    stage64(0, 0);
    Frag a0, a1;
    a0.q[0] = *reinterpret_cast<const uint4*>(aRow);
    a0.q[1] = *reinterpret_cast<const uint4*>(aRow + 16);
    a1.q[0] = *reinterpret_cast<const uint4*>(aRow + 32);
    a1.q[1] = *reinterpret_cast<const uint4*>(aRow + 48);
    wait_async0();
    __syncthreads();

    for (int k0 = 0; k0 < H_; k0 += 64) {
        const int cur = (k0 >> 6) & 1;
        const int nxt = cur ^ 1;
        Frag a0n, a1n;
        if (k0 + 64 < H_) {
            stage64(nxt, k0 + 64);                     // overlap: next B slab -> LDS
            a0n.q[0] = *reinterpret_cast<const uint4*>(aRow + k0 + 64);
            a0n.q[1] = *reinterpret_cast<const uint4*>(aRow + k0 + 80);
            a1n.q[0] = *reinterpret_cast<const uint4*>(aRow + k0 + 96);
            a1n.q[1] = *reinterpret_cast<const uint4*>(aRow + k0 + 112);
        }
        Frag b4[4];
#pragma unroll
        for (int j = 0; j < 4; ++j)
            b4[j] = *reinterpret_cast<const Frag*>(&sB[cur][(j * 32 + lane) * 16]);
#pragma unroll
        for (int j = 0; j < 4; ++j)
            acc[j] = __builtin_amdgcn_wmma_f32_16x16x32_bf16(
                false, a0.v, false, b4[j].v, (short)0, acc[j], false, false);
#pragma unroll
        for (int j = 0; j < 4; ++j)
            b4[j] = *reinterpret_cast<const Frag*>(&sB[cur][2048 + (j * 32 + lane) * 16]);
#pragma unroll
        for (int j = 0; j < 4; ++j)
            acc[j] = __builtin_amdgcn_wmma_f32_16x16x32_bf16(
                false, a1.v, false, b4[j].v, (short)0, acc[j], false, false);
        wait_async0();
        __syncthreads();
        a0 = a0n; a1 = a1n;
    }

#pragma unroll
    for (int j = 0; j < 4; ++j) {
        const int col = n0 + 16 * j + l15;
#pragma unroll
        for (int i = 0; i < 8; ++i) {
            const int row = m0 + i + (hi ? 8 : 0);   // C/D layout: VGPR i -> rows i / i+8
            Xg[(size_t)row * G4H + col] = acc[j][i];
        }
    }
}

// ---------------- grid-wide barrier ----------------
// Cluster HW barrier first (NOP when not dispatched as a cluster), then a
// monotonic atomic-counter barrier that is always correct for 32 resident WGs.
__device__ __forceinline__ void grid_barrier(unsigned int* bar, unsigned int target) {
    __threadfence();
    __syncthreads();
    __builtin_amdgcn_s_cluster_barrier();
    if (threadIdx.x == 0) {
        __hip_atomic_fetch_add(bar, 1u, __ATOMIC_RELEASE, __HIP_MEMORY_SCOPE_AGENT);
        while (__hip_atomic_load(bar, __ATOMIC_ACQUIRE, __HIP_MEMORY_SCOPE_AGENT) < target)
            __builtin_amdgcn_s_sleep(1);
    }
    __syncthreads();
}

// ---------------- persistent recurrent kernel: one launch per layer ----------------
// 32 WGs x 4 waves = 128 waves; wave (r,cc) owns h-tile [16r:16r+16, 16cc:16cc+16].
// Each wave's 64KB slice of bf16 Whh (4 gates x 16 cols x K=512) is async-copied
// into LDS once; every timestep then runs entirely out of LDS (B) + registers
// (c-state, A preloaded for the whole K-loop). One grid barrier per step.
__global__ __launch_bounds__(REC_TPB, 1)
void k_lstm_rec(const float* __restrict__ Xg,            // [M,2048] f32, row=(b*T+t)
                const unsigned short* __restrict__ Whh,  // [2048,512] bf16
                unsigned short* __restrict__ h0buf,      // [64,512] bf16 ping
                unsigned short* __restrict__ h1buf,      // [64,512] bf16 pong
                unsigned short* __restrict__ outBf,      // [B,T,512] bf16 (next layer A)
                float* __restrict__ outF,                // [B,T,512] f32 (final output)
                int writeF32,
                unsigned int* __restrict__ bar,
                unsigned int target0) {
    extern __shared__ unsigned short sW[];               // 4 waves x 64KB = 256KB

    const int lane = threadIdx.x & 31;
    const int wave = threadIdx.x >> 5;                   // 0..3
    const int wid  = (blockIdx.x << 2) + wave;           // 0..127
    const int r    = wid >> 5;     // 0..3   (64 rows / 16)
    const int cc   = wid & 31;     // 0..31  (512 cols / 16)
    const int l15  = lane & 15;
    const int hi   = lane >> 4;
    const int akoff = hi ? 8 : 0;
    const int bkoff = hi ? 8 : 0;

    unsigned int target = target0;

    // ---- cache this wave's Whh working set in LDS (async DMA) ----
    // ushort index = wave*32768 + (j*16+kk)*512 + lane*16  (32B per lane/fragment)
    {
        const unsigned ldsBase =
            (unsigned)(uintptr_t)(&sW[0]) + (unsigned)(wave * 65536);
#pragma unroll
        for (int j = 0; j < 4; ++j) {
            const unsigned short* src =
                Whh + (size_t)(j * H_ + cc * 16 + l15) * H_ + bkoff;
            for (int kk = 0; kk < 16; ++kk) {
                unsigned lds = ldsBase + (unsigned)(((j * 16 + kk) * 32 + lane) * 32);
                async_g2l_b128(lds,      src + kk * 32);
                async_g2l_b128(lds + 16, src + kk * 32 + 16);
            }
        }
    }

    // ---- zero h for t=0 ----
    {
        const int col = cc * 16 + l15;
#pragma unroll
        for (int i = 0; i < 8; ++i)
            h0buf[(r * 16 + i + hi * 8) * H_ + col] = 0;
    }
    wait_async0();
    grid_barrier(bar, target); target += NWG_REC;

    v8f cstate;
#pragma unroll
    for (int i = 0; i < 8; ++i) cstate[i] = 0.0f;

    const unsigned short* sWav = &sW[wave * 32768];

    for (int t = 0; t < T_; ++t) {
        const unsigned short* hcur = (t & 1) ? h1buf : h0buf;
        unsigned short*       hnxt = (t & 1) ? h0buf : h1buf;

        // init accumulators from precomputed input projection (gate order i,f,g,o)
        v8f acc[4];
#pragma unroll
        for (int j = 0; j < 4; ++j) {
            const int n = j * H_ + cc * 16 + l15;
#pragma unroll
            for (int i = 0; i < 8; ++i) {
                const int b = r * 16 + i + hi * 8;
                acc[j][i] = Xg[((size_t)b * T_ + t) * G4H + n];
            }
        }

        // preload the entire A (h_t) K-strip for this wave into registers
        Frag afr[16];
        const unsigned short* aRow = hcur + (r * 16 + l15) * H_ + akoff;
#pragma unroll
        for (int kk = 0; kk < 16; ++kk) {
            afr[kk].q[0] = *reinterpret_cast<const uint4*>(aRow + kk * 32);
            afr[kk].q[1] = *reinterpret_cast<const uint4*>(aRow + kk * 32 + 16);
        }

        // g += h_t * Whh^T ; B entirely from LDS, batched 4-at-a-time before WMMAs
#pragma unroll
        for (int kk = 0; kk < 16; ++kk) {
            Frag b4[4];
#pragma unroll
            for (int j = 0; j < 4; ++j)
                b4[j] = *reinterpret_cast<const Frag*>(
                    &sWav[((j * 16 + kk) * 32 + lane) * 16]);
#pragma unroll
            for (int j = 0; j < 4; ++j)
                acc[j] = __builtin_amdgcn_wmma_f32_16x16x32_bf16(
                    false, afr[kk].v, false, b4[j].v, (short)0, acc[j], false, false);
        }

        // gates -> (c,h); write h (bf16 for next step / next layer, f32 final out)
        const int col = cc * 16 + l15;
#pragma unroll
        for (int i = 0; i < 8; ++i) {
            const int b = r * 16 + i + hi * 8;
            float iv = sigmoidf_(acc[0][i]);
            float fv = sigmoidf_(acc[1][i]);
            float gv = tanhf(acc[2][i]);
            float ov = sigmoidf_(acc[3][i]);
            float cv = fv * cstate[i] + iv * gv;
            cstate[i] = cv;
            float hv = ov * tanhf(cv);
            unsigned short hb = f2bf(hv);
            hnxt[b * H_ + col] = hb;
            size_t oidx = ((size_t)b * T_ + t) * H_ + col;
            outBf[oidx] = hb;
            if (writeF32) outF[oidx] = hv;
        }

        grid_barrier(bar, target); target += NWG_REC;   // h_{t+1} visible before next step
    }
}

// ---------------- host side ----------------
extern "C" void kernel_launch(void* const* d_in, const int* in_sizes, int n_in,
                              void* d_out, int out_size, void* d_ws, size_t ws_size,
                              hipStream_t stream) {
    const float* x   = (const float*)d_in[0];   // [B,T,H]
    const float* wih = (const float*)d_in[1];   // [L,4H,H]
    const float* whh = (const float*)d_in[2];   // [L,4H,H]
    const float* bih = (const float*)d_in[3];   // [L,4H]
    const float* bhh = (const float*)d_in[4];   // [L,4H]
    float* out = (float*)d_out;                 // [B,T,H]

    char* ws = (char*)d_ws;
    size_t off = 0;
    auto carve = [&](size_t bytes) {
        void* p = ws + off;
        off = (off + bytes + 255) & ~(size_t)255;
        return p;
    };
    float*          xg    = (float*)carve((size_t)M_ * G4H * sizeof(float));       // 512 MB
    unsigned short* xA    = (unsigned short*)carve((size_t)M_ * H_ * 2);           // 64 MB
    unsigned short* xB    = (unsigned short*)carve((size_t)M_ * H_ * 2);           // 64 MB
    unsigned short* wihb  = (unsigned short*)carve((size_t)L_ * G4H * H_ * 2);
    unsigned short* whhb  = (unsigned short*)carve((size_t)L_ * G4H * H_ * 2);
    float*          biasc = (float*)carve((size_t)L_ * G4H * sizeof(float));
    unsigned short* h0b   = (unsigned short*)carve((size_t)B_ * H_ * 2);
    unsigned short* h1b   = (unsigned short*)carve((size_t)B_ * H_ * 2);
    unsigned int*   bar   = (unsigned int*)carve(256);

    // allow 256KB dynamic LDS for the recurrent kernel
    (void)hipFuncSetAttribute((const void*)k_lstm_rec,
                              hipFuncAttributeMaxDynamicSharedMemorySize, 262144);

    // bf16 conversions (x once; weights once) + combined bias + barrier reset
    k_convert_bf16<<<2048, 256, 0, stream>>>(x,   xA,   M_ * H_);
    k_convert_bf16<<<2048, 256, 0, stream>>>(wih, wihb, L_ * G4H * H_);
    k_convert_bf16<<<2048, 256, 0, stream>>>(whh, whhb, L_ * G4H * H_);
    k_bias_bar<<<32, 256, 0, stream>>>(bih, bhh, biasc, L_ * G4H, bar);

    unsigned short* inb  = xA;
    unsigned short* outb = xB;
    for (int l = 0; l < L_; ++l) {
        dim3 grid(M_ / 128, G4H / 64);   // 512 x 32 blocks of 256 threads
        k_gemm_xproj<<<grid, 256, 0, stream>>>(
            inb, wihb + (size_t)l * G4H * H_, biasc + (size_t)l * G4H, xg);

        // barriers per layer: 1 (init) + T_ (steps); monotonic counter target base
        unsigned int t0 = (unsigned int)NWG_REC * ((unsigned int)l * (T_ + 1u) + 1u);
        k_lstm_rec<<<NWG_REC, REC_TPB, 262144, stream>>>(
            xg, whhb + (size_t)l * G4H * H_, h0b, h1b, outb, out,
            (l == L_ - 1) ? 1 : 0, bar, t0);

        unsigned short* tmp = inb; inb = outb; outb = tmp;
    }
}
